// SableNetwork_2379411882668
// MI455X (gfx1250) — matrix operand order repaired and checked by reading, not compiled, not run on hardware
//
#include <hip/hip_runtime.h>
#include <hip/hip_bf16.h>
#include <math.h>

// ---- problem constants (match reference) ----
#define BB      4
#define SS      1024
#define OBSD    128
#define DD      512
#define HH      8
#define DHH     64
#define NBLK    2
#define ACTN    64
#define NAGENT  8
#define BSROWS  (BB * SS)          // 4096
#define ACT_KP  96                 // A+1=65 padded to multiple of 32

typedef __attribute__((ext_vector_type(16))) _Float16 v16h;
typedef __attribute__((ext_vector_type(8)))  _Float16 v8h;
typedef __attribute__((ext_vector_type(8)))  float    v8f;

__device__ __forceinline__ v16h cat8(v8h lo, v8h hi) {
    return __builtin_shufflevector(lo, hi, 0,1,2,3,4,5,6,7,8,9,10,11,12,13,14,15);
}
__device__ __forceinline__ v8f wmma_f16(v16h a, v16h b, v8f c) {
    return __builtin_amdgcn_wmma_f32_16x16x32_f16(false, a, false, b, (short)0, c, false, false);
}
__device__ __forceinline__ float gelu_f(float x) {
    float x3 = x * x * x;
    return 0.5f * x * (1.0f + tanhf(0.7978845608028654f * (x + 0.044715f * x3)));
}
__device__ __forceinline__ float silu_f(float x) { return x / (1.0f + expf(-x)); }

// ---- epilogue modes ----
enum {
    EP_F32F16 = 0,   // write f32 + f16
    EP_F16    = 1,   // write f16 only
    EP_VTRANS = 2,   // write f16 per-head transposed Vt[b,h,dh,s]
    EP_GELU   = 3,   // gelu -> f32 (+f16 if provided)
    EP_GELU_B = 4,   // gelu(acc+bias) -> f32
    EP_SILUMUL= 5,   // silu(aux)*acc -> f16
    EP_ADDRES = 6,   // acc+aux -> f32
    EP_MASKB  = 7,   // (acc+bias) masked by legal -> f32
    EP_F32    = 8    // f32 only
};

// ---------------------------------------------------------------------------
// Repack f32 weight W[K,N] into WMMA B-operand tile layout (f16).
// Block b = (ntile*(Kpad/32) + ktile), 512 halves/block.
// Within block: half index = lane*16 + j ; K = ktile*32 + (lane<16?0:16) + j ;
//               N = ntile*16 + lane%16.  Pads K rows >= Kreal with zeros.
// ---------------------------------------------------------------------------
__global__ void k_repack(const float* __restrict__ W, _Float16* __restrict__ Wp,
                         int Kreal, int Kpad, int N) {
    int tid = blockIdx.x * blockDim.x + threadIdx.x;
    int total = Kpad * N;
    if (tid >= total) return;
    int blk    = tid >> 9;
    int within = tid & 511;
    int lane   = within >> 4;
    int j      = within & 15;
    int ktiles = Kpad >> 5;
    int ktile  = blk % ktiles;
    int ntile  = blk / ktiles;
    int krow   = (ktile << 5) + ((lane < 16) ? 0 : 16) + j;
    int n      = (ntile << 4) + (lane & 15);
    float v = (krow < Kreal) ? W[(size_t)krow * N + n] : 0.0f;
    Wp[tid] = (_Float16)v;
}

// ---------------------------------------------------------------------------
// RMSNorm over last dim L (one wave per row), optional f32 / f16 outputs.
// ---------------------------------------------------------------------------
__global__ void k_rmsnorm(const float* __restrict__ X, const float* __restrict__ g,
                          float* __restrict__ outF, _Float16* __restrict__ outH, int L) {
    int row  = blockIdx.x;
    int lane = threadIdx.x;
    const float* xr = X + (size_t)row * L;
    float ss = 0.0f;
    for (int c = lane; c < L; c += 32) { float v = xr[c]; ss += v * v; }
#pragma unroll
    for (int m = 16; m >= 1; m >>= 1) ss += __shfl_xor(ss, m, 32);
    float inv = rsqrtf(ss / (float)L + 1e-6f);
    for (int c = lane; c < L; c += 32) {
        float v = xr[c] * g[c] * inv;
        if (outF) outF[(size_t)row * L + c] = v;
        if (outH) outH[(size_t)row * L + c] = (_Float16)v;
    }
}

// ---------------------------------------------------------------------------
// Build shifted one-hot action tokens (f16, padded to ACT_KP cols).
// ---------------------------------------------------------------------------
__global__ void k_build_act(const int* __restrict__ action, _Float16* __restrict__ acth) {
    int row = blockIdx.x * blockDim.x + threadIdx.x;
    if (row >= BSROWS) return;
    int s = row & (SS - 1);
    _Float16* o = acth + (size_t)row * ACT_KP;
    for (int c = 0; c < ACT_KP; c++) o[c] = (_Float16)0.0f;
    if ((s % NAGENT) == 0) o[0] = (_Float16)1.0f;
    else                   o[1 + action[row - 1]] = (_Float16)1.0f;
}

// ---------------------------------------------------------------------------
// WMMA GEMM: one wave -> 16(M) x 64(N) tile.  A: row-major f16 [M,K].
// Wp: packed B tiles, double-buffered through LDS with CDNA5 async-to-LDS
// copies (ASYNCcnt) so weight fetch latency overlaps the matrix pipe.
// ---------------------------------------------------------------------------
__global__ void k_gemm(const _Float16* __restrict__ A, const _Float16* __restrict__ Wp,
                       int K, int N, int mode,
                       const float* __restrict__ aux, const float* __restrict__ bias,
                       const unsigned char* __restrict__ legal,
                       float* __restrict__ outF, _Float16* __restrict__ outH) {
    __shared__ __align__(16) _Float16 ldsB[2][2048];   // 2 x 4KB: 4 n-subtiles x 512 halves
    int m0   = blockIdx.y << 4;
    int n0   = blockIdx.x << 6;
    int lane = threadIdx.x;
    int lmod = lane & 15;
    int lhi  = (lane < 16) ? 0 : 1;
    int row  = m0 + lmod;
    const _Float16* arow = A + (size_t)row * K;
    v8f acc0 = {}, acc1 = {}, acc2 = {}, acc3 = {};
    int ktiles = K >> 5;
    int nt0    = n0 >> 4;

    // async copy of one k-step's 4 weight blocks into LDS buffer bufi
    auto issueB = [&](int kt, int bufi) {
#pragma unroll
        for (int ns = 0; ns < 4; ns++) {
            unsigned dst = (unsigned)(uintptr_t)(&ldsB[bufi][ns * 512 + lane * 16]);
            const _Float16* gp = Wp + ((size_t)(nt0 + ns) * ktiles + kt) * 512 + lane * 16;
            // IOFFSET applies to both LDS and global address (ISA 08 §4.4)
            asm volatile(
                "global_load_async_to_lds_b128 %0, %1, off\n\t"
                "global_load_async_to_lds_b128 %0, %1, off offset:16"
                :: "v"(dst), "v"(gp) : "memory");
        }
    };

    issueB(0, 0);
    for (int kt = 0; kt < ktiles; kt++) {
        bool more = (kt + 1 < ktiles);
        if (more) issueB(kt + 1, (kt + 1) & 1);
        // A operand: lanes 0-15 hold K=[0..15]; lanes 16-31 hold K=[16..31]
        int off = (kt << 5) + (lhi ? 16 : 0);
        v8h r1 = *(const v8h*)(arow + off);
        v8h r2 = *(const v8h*)(arow + off + 8);
        v16h a = cat8(r1, r2);
        if (more) __builtin_prefetch(arow + off + 32, 0, 1);
        // current buffer's 8 in-order async loads are complete when <=8 remain
        if (more) asm volatile("s_wait_asynccnt 0x8" ::: "memory");
        else      asm volatile("s_wait_asynccnt 0x0" ::: "memory");
        const _Float16* lb = &ldsB[kt & 1][lane * 16];
        v16h b0 = *(const v16h*)(lb);
        v16h b1 = *(const v16h*)(lb + 512);
        v16h b2 = *(const v16h*)(lb + 1024);
        v16h b3 = *(const v16h*)(lb + 1536);
        acc0 = wmma_f16(a, b0, acc0);
        acc1 = wmma_f16(a, b1, acc1);
        acc2 = wmma_f16(a, b2, acc2);
        acc3 = wmma_f16(a, b3, acc3);
    }
    v8f accs[4] = {acc0, acc1, acc2, acc3};
#pragma unroll
    for (int t = 0; t < 4; t++) {
#pragma unroll
        for (int r = 0; r < 8; r++) {
            int orow = m0 + r + (lhi ? 8 : 0);       // C layout: VGPR r -> M=r (lo lanes) / 8+r (hi)
            int ocol = n0 + t * 16 + lmod;
            size_t idx = (size_t)orow * N + ocol;
            float v = accs[t][r];
            switch (mode) {
                case EP_F32F16: outF[idx] = v; outH[idx] = (_Float16)v; break;
                case EP_F16:    outH[idx] = (_Float16)v; break;
                case EP_F32:    outF[idx] = v; break;
                case EP_VTRANS: {
                    int b = orow >> 10, s = orow & 1023, h = ocol >> 6, dh = ocol & 63;
                    outH[((size_t)(b * HH + h) * DHH + dh) * SS + s] = (_Float16)v;
                } break;
                case EP_GELU: {
                    float gv = gelu_f(v);
                    outF[idx] = gv;
                    if (outH) outH[idx] = (_Float16)gv;
                } break;
                case EP_GELU_B: outF[idx] = gelu_f(v + bias[ocol]); break;
                case EP_SILUMUL: outH[idx] = (_Float16)(silu_f(aux[idx]) * v); break;
                case EP_ADDRES:  outF[idx] = v + aux[idx]; break;
                case EP_MASKB: {
                    float o = v + bias[ocol];
                    outF[idx] = legal[idx] ? o : -3.4028234663852886e38f;
                } break;
            }
        }
    }
}

// ---------------------------------------------------------------------------
// Fused multi-scale retention, one wave per (b, h, 16 q-rows).
// Computes S^T = K_tile . Q^T so the decayed scores land (after a purely
// intra-lane f32->f16 pack) directly in WMMA A-layout for out += S_dec . V.
// ---------------------------------------------------------------------------
__global__ void k_retention(const _Float16* __restrict__ Qh, const _Float16* __restrict__ Kh,
                            const _Float16* __restrict__ Vt, _Float16* __restrict__ Rh,
                            int causal) {
    int qt   = blockIdx.x;
    int bh   = blockIdx.y;
    int b    = bh >> 3;
    int h    = bh & 7;
    int lane = threadIdx.x;
    int lmod = lane & 15;
    int lhi  = (lane < 16) ? 0 : 1;
    float gamma = 1.0f - exp2f(-5.0f - (float)h);
    float l2g   = log2f(gamma);                    // negative
    const float scale = 0.125f;                    // 1/sqrt(64)
    // Q as B operand (N=q on lanes, K=dh on halves), two dh chunks of 32
    int qrow = b * SS + qt * 16 + lmod;
    const _Float16* qp = Qh + (size_t)qrow * DD + h * DHH;
    v16h qb0 = *(const v16h*)(qp +      (lhi ? 16 : 0));
    v16h qb1 = *(const v16h*)(qp + 32 + (lhi ? 16 : 0));
    v8f acc[4] = {{}, {}, {}, {}};
    int kvTiles = causal ? (qt / 2 + 1) : (SS / 32);
    int qcol = qt * 16 + lmod;
    const _Float16* vbase = Vt + (size_t)bh * DHH * SS;
    for (int kt = 0; kt < kvTiles; kt++) {
        int kv0 = kt * 32;
        const _Float16* k0p = Kh + (size_t)(b * SS + kv0 + lmod) * DD + h * DHH;
        const _Float16* k1p = Kh + (size_t)(b * SS + kv0 + 16 + lmod) * DD + h * DHH;
        int aoff = (lhi ? 16 : 0);
        v16h ka00 = cat8(*(const v8h*)(k0p + aoff),      *(const v8h*)(k0p + aoff + 8));
        v16h ka01 = cat8(*(const v8h*)(k0p + 32 + aoff), *(const v8h*)(k0p + 32 + aoff + 8));
        v16h ka10 = cat8(*(const v8h*)(k1p + aoff),      *(const v8h*)(k1p + aoff + 8));
        v16h ka11 = cat8(*(const v8h*)(k1p + 32 + aoff), *(const v8h*)(k1p + 32 + aoff + 8));
        v8f c0 = {}, c1 = {};
        c0 = wmma_f16(ka00, qb0, c0); c0 = wmma_f16(ka01, qb1, c0);
        c1 = wmma_f16(ka10, qb0, c1); c1 = wmma_f16(ka11, qb1, c1);
        // decay + scale + pack (intra-lane): A halves 0..7 <- c0, 8..15 <- c1
        v16h sa;
#pragma unroll
        for (int r = 0; r < 8; r++) {
            int kvA = kv0 + r + (lhi ? 8 : 0);
            int kvB = kvA + 16;
            float dA = (float)(qcol - kvA);
            float dB = (float)(qcol - kvB);
            float wA, wB;
            if (causal) {
                wA = (dA >= 0.0f) ? exp2f(dA * l2g) : 0.0f;
                wB = (dB >= 0.0f) ? exp2f(dB * l2g) : 0.0f;
            } else {
                wA = exp2f(fabsf(dA) * l2g);
                wB = exp2f(fabsf(dB) * l2g);
            }
            sa[r]     = (_Float16)(c0[r] * scale * wA);
            sa[r + 8] = (_Float16)(c1[r] * scale * wB);
        }
        const _Float16* vp = vbase + kv0 + (lhi ? 8 : 0);
#pragma unroll
        for (int t = 0; t < 4; t++) {
            v16h vb = cat8(*(const v8h*)(vp + (size_t)(t * 16 + lmod) * SS), *(const v8h*)(vp + (size_t)(t * 16 + lmod) * SS + 16));
            acc[t] = wmma_f16(sa, vb, acc[t]);
        }
    }
#pragma unroll
    for (int t = 0; t < 4; t++) {
#pragma unroll
        for (int r = 0; r < 8; r++) {
            int orow = b * SS + qt * 16 + r + (lhi ? 8 : 0);
            int ocol = h * DHH + t * 16 + lmod;
            Rh[(size_t)orow * DD + ocol] = (_Float16)acc[t][r];
        }
    }
}

// ---------------------------------------------------------------------------
// Host side
// ---------------------------------------------------------------------------
extern "C" void kernel_launch(void* const* d_in, const int* in_sizes, int n_in,
                              void* d_out, int out_size, void* d_ws, size_t ws_size,
                              hipStream_t stream) {
    if (n_in < 58) return;
    const float* obs;
    const int* action;
    const unsigned char* legal;
    const float* prm[55];
    int pbase;
    if (in_sizes[0] == BSROWS * OBSD) {          // insertion order: obs, params..., action, legal
        obs    = (const float*)d_in[0];
        pbase  = 1;
        action = (const int*)d_in[56];
        legal  = (const unsigned char*)d_in[57];
    } else {                                      // sorted keys: action, legal, obs, params...
        action = (const int*)d_in[0];
        legal  = (const unsigned char*)d_in[1];
        obs    = (const float*)d_in[2];
        pbase  = 3;
    }
    for (int i = 0; i < 55; i++) prm[i] = (const float*)d_in[pbase + i];

    // params subtree in JAX sorted-key order
    const float* dec_W_act = prm[0];
    const float *dW1[2], *dW2[2], *dWg[2], *dln1[2], *dln2[2], *dln3[2];
    const float *r1Wk[2], *r1Wo[2], *r1Wq[2], *r1Wv[2];
    const float *r2Wk[2], *r2Wo[2], *r2Wq[2], *r2Wv[2];
    for (int i = 0; i < 2; i++) {
        const float* const* p = prm + 1 + i * 14;
        dW1[i]=p[0]; dW2[i]=p[1]; dWg[i]=p[2]; dln1[i]=p[3]; dln2[i]=p[4]; dln3[i]=p[5];
        r1Wk[i]=p[6]; r1Wo[i]=p[7]; r1Wq[i]=p[8]; r1Wv[i]=p[9];
        r2Wk[i]=p[10]; r2Wo[i]=p[11]; r2Wq[i]=p[12]; r2Wv[i]=p[13];
    }
    const float* dec_ln    = prm[29];
    const float* enc_W_obs = prm[30];
    const float *eW1[2], *eW2[2], *eWg[2], *eln1[2], *eln2[2];
    const float *eWk[2], *eWo[2], *eWq[2], *eWv[2];
    for (int i = 0; i < 2; i++) {
        const float* const* p = prm + 31 + i * 9;
        eW1[i]=p[0]; eW2[i]=p[1]; eWg[i]=p[2]; eln1[i]=p[3]; eln2[i]=p[4];
        eWk[i]=p[5]; eWo[i]=p[6]; eWq[i]=p[7]; eWv[i]=p[8];
    }
    const float* enc_obs_ln = prm[49];
    const float* head_W1 = prm[50];
    const float* head_W2 = prm[51];
    const float* head_b1 = prm[52];
    const float* head_b2 = prm[53];
    const float* head_ln = prm[54];

    // bump allocator on d_ws
    char* wsb = (char*)d_ws;
    size_t off = 0;
    auto alloc = [&](size_t bytes) -> void* {
        void* p = wsb + off;
        off += (bytes + 255) & ~(size_t)255;
        return p;
    };
    auto pack = [&](const float* W, int Kreal, int Kpad, int N) -> _Float16* {
        _Float16* p = (_Float16*)alloc((size_t)Kpad * N * sizeof(_Float16));
        int total = Kpad * N;
        hipLaunchKernelGGL(k_repack, dim3((total + 255) / 256), dim3(256), 0, stream,
                           W, p, Kreal, Kpad, N);
        return p;
    };

    // pack all weights once per call
    _Float16* pObs = pack(enc_W_obs, OBSD, OBSD, DD);
    _Float16* pAct = pack(dec_W_act, ACTN + 1, ACT_KP, DD);
    _Float16 *peWq[2], *peWk[2], *peWv[2], *peWo[2], *peWg[2], *peW1[2], *peW2[2];
    _Float16 *p1Wq[2], *p1Wk[2], *p1Wv[2], *p1Wo[2];
    _Float16 *p2Wq[2], *p2Wk[2], *p2Wv[2], *p2Wo[2];
    _Float16 *pdWg[2], *pdW1[2], *pdW2[2];
    for (int i = 0; i < 2; i++) {
        peWq[i]=pack(eWq[i],DD,DD,DD); peWk[i]=pack(eWk[i],DD,DD,DD);
        peWv[i]=pack(eWv[i],DD,DD,DD); peWo[i]=pack(eWo[i],DD,DD,DD);
        peWg[i]=pack(eWg[i],DD,DD,DD); peW1[i]=pack(eW1[i],DD,DD,DD); peW2[i]=pack(eW2[i],DD,DD,DD);
        p1Wq[i]=pack(r1Wq[i],DD,DD,DD); p1Wk[i]=pack(r1Wk[i],DD,DD,DD);
        p1Wv[i]=pack(r1Wv[i],DD,DD,DD); p1Wo[i]=pack(r1Wo[i],DD,DD,DD);
        p2Wq[i]=pack(r2Wq[i],DD,DD,DD); p2Wk[i]=pack(r2Wk[i],DD,DD,DD);
        p2Wv[i]=pack(r2Wv[i],DD,DD,DD); p2Wo[i]=pack(r2Wo[i],DD,DD,DD);
        pdWg[i]=pack(dWg[i],DD,DD,DD); pdW1[i]=pack(dW1[i],DD,DD,DD); pdW2[i]=pack(dW2[i],DD,DD,DD);
    }
    _Float16* pH1 = pack(head_W1, DD, DD, DD);
    _Float16* pH2 = pack(head_W2, DD, DD, ACTN);

    // activation buffers
    size_t fsz = (size_t)BSROWS * DD * sizeof(float);
    size_t hsz = (size_t)BSROWS * DD * sizeof(_Float16);
    float    *Af  = (float*)alloc(fsz), *Bf = (float*)alloc(fsz);
    float    *T1f = (float*)alloc(fsz), *T2f = (float*)alloc(fsz);
    _Float16 *Ah  = (_Float16*)alloc(hsz), *Bh = (_Float16*)alloc(hsz), *Ch = (_Float16*)alloc(hsz);
    _Float16 *Qh  = (_Float16*)alloc(hsz), *Kx = (_Float16*)alloc(hsz);
    _Float16 *Vt  = (_Float16*)alloc(hsz), *Rh = (_Float16*)alloc(hsz);
    _Float16 *Sh  = (_Float16*)alloc(hsz), *Hh = (_Float16*)alloc(hsz);
    _Float16 *obsh = (_Float16*)alloc((size_t)BSROWS * OBSD * sizeof(_Float16));
    _Float16 *acth = (_Float16*)alloc((size_t)BSROWS * ACT_KP * sizeof(_Float16));
    (void)ws_size; (void)out_size;

    auto gemm = [&](const _Float16* A, const _Float16* Wp, int K, int N, int mode,
                    const float* aux, const float* bias, const unsigned char* lg,
                    float* oF, _Float16* oH) {
        hipLaunchKernelGGL(k_gemm, dim3(N / 64, BSROWS / 16), dim3(32), 0, stream,
                           A, Wp, K, N, mode, aux, bias, lg, oF, oH);
    };
    auto rms = [&](const float* X, const float* g, float* oF, _Float16* oH, int L) {
        hipLaunchKernelGGL(k_rmsnorm, dim3(BSROWS), dim3(32), 0, stream, X, g, oF, oH, L);
    };
    auto ret = [&](const _Float16* Q, const _Float16* Kk, const _Float16* V, _Float16* R, int causal) {
        hipLaunchKernelGGL(k_retention, dim3(SS / 16, BB * HH), dim3(32), 0, stream,
                           Q, Kk, V, R, causal);
    };

    // ---- encoder ----
    rms(obs, enc_obs_ln, nullptr, obsh, OBSD);
    gemm(obsh, pObs, OBSD, DD, EP_GELU, nullptr, nullptr, nullptr, Af, Ah);
    for (int i = 0; i < NBLK; i++) {
        gemm(Ah, peWq[i], DD, DD, EP_F16,    nullptr, nullptr, nullptr, nullptr, Qh);
        gemm(Ah, peWk[i], DD, DD, EP_F16,    nullptr, nullptr, nullptr, nullptr, Kx);
        gemm(Ah, peWv[i], DD, DD, EP_VTRANS, nullptr, nullptr, nullptr, nullptr, Vt);
        ret(Qh, Kx, Vt, Rh, 0);
        gemm(Rh, peWo[i], DD, DD, EP_ADDRES, Af, nullptr, nullptr, T1f, nullptr);
        rms(T1f, eln1[i], Af, Ah, DD);
        gemm(Ah, peWg[i], DD, DD, EP_F32,     nullptr, nullptr, nullptr, T2f, nullptr);
        gemm(Ah, peW1[i], DD, DD, EP_SILUMUL, T2f, nullptr, nullptr, nullptr, Sh);
        gemm(Sh, peW2[i], DD, DD, EP_ADDRES,  Af, nullptr, nullptr, T1f, nullptr);
        rms(T1f, eln2[i], Af, Ah, DD);
    }
    // Af/Ah now hold obs_rep (f32 residual + f16 projections input)

    // ---- decoder ----
    hipLaunchKernelGGL(k_build_act, dim3((BSROWS + 255) / 256), dim3(256), 0, stream, action, acth);
    gemm(acth, pAct, ACT_KP, DD, EP_GELU, nullptr, nullptr, nullptr, T1f, nullptr);
    rms(T1f, dec_ln, Bf, Bh, DD);
    for (int i = 0; i < NBLK; i++) {
        // causal self-retention on y
        gemm(Bh, p1Wq[i], DD, DD, EP_F16,    nullptr, nullptr, nullptr, nullptr, Qh);
        gemm(Bh, p1Wk[i], DD, DD, EP_F16,    nullptr, nullptr, nullptr, nullptr, Kx);
        gemm(Bh, p1Wv[i], DD, DD, EP_VTRANS, nullptr, nullptr, nullptr, nullptr, Vt);
        ret(Qh, Kx, Vt, Rh, 1);
        gemm(Rh, p1Wo[i], DD, DD, EP_ADDRES, Bf, nullptr, nullptr, T1f, nullptr);
        rms(T1f, dln1[i], nullptr, Ch, DD);
        // cross retention: q from obs_rep, k/v from x
        gemm(Ah, p2Wq[i], DD, DD, EP_F16,    nullptr, nullptr, nullptr, nullptr, Qh);
        gemm(Ch, p2Wk[i], DD, DD, EP_F16,    nullptr, nullptr, nullptr, nullptr, Kx);
        gemm(Ch, p2Wv[i], DD, DD, EP_VTRANS, nullptr, nullptr, nullptr, nullptr, Vt);
        ret(Qh, Kx, Vt, Rh, 1);
        gemm(Rh, p2Wo[i], DD, DD, EP_ADDRES, Af, nullptr, nullptr, T1f, nullptr);
        rms(T1f, dln2[i], Bf, Bh, DD);
        // swiglu
        gemm(Bh, pdWg[i], DD, DD, EP_F32,     nullptr, nullptr, nullptr, T2f, nullptr);
        gemm(Bh, pdW1[i], DD, DD, EP_SILUMUL, T2f, nullptr, nullptr, nullptr, Sh);
        gemm(Sh, pdW2[i], DD, DD, EP_ADDRES,  Bf, nullptr, nullptr, T1f, nullptr);
        rms(T1f, dln3[i], Bf, Bh, DD);
    }

    // ---- head ----
    gemm(Bh, pH1, DD, DD, EP_GELU_B, nullptr, head_b1, nullptr, T1f, nullptr);
    rms(T1f, head_ln, nullptr, Hh, DD);
    gemm(Hh, pH2, DD, ACTN, EP_MASKB, nullptr, head_b2, legal, (float*)d_out, nullptr);
}
